// BEV_DFA_layer_51376398795123
// MI455X (gfx1250) — compile-verified, matrix operand order checked by reading, not run on hardware
//
#include <hip/hip_runtime.h>

typedef _Float16 v16h __attribute__((ext_vector_type(16)));
typedef _Float16 v8h  __attribute__((ext_vector_type(8)));
typedef float    v8f  __attribute__((ext_vector_type(8)));

#define KDIM 256

// ---------------------------------------------------------------------------
// Pack a (K=256 x N) f32 weight matrix into WMMA B-fragment layout (f16).
// ISA 7.12.2 (16-bit operand, wave32): lane holds one column (col = lane&15);
// lanes<16 hold K {kt*32+0..7, kt*32+16..23}, lanes>=16 hold {+8..15,+24..31}.
// Packed element index: (((nt*8 + kt)*32 + lane)*16 + j)
// ---------------------------------------------------------------------------
__global__ void pack_w_kernel(const float* __restrict__ W, _Float16* __restrict__ Wp, int N) {
    int t = blockIdx.x * blockDim.x + threadIdx.x;
    int total = (N >> 4) * 8 * 32;
    if (t >= total) return;
    int lane   = t & 31;
    int kt     = (t >> 5) & 7;
    int nt     = t >> 8;
    int col    = lane & 15;
    int halfid = lane >> 4;
    int n      = nt * 16 + col;
    _Float16* dst = Wp + (size_t)t * 16;
#pragma unroll
    for (int j = 0; j < 8; ++j) {
        int k0 = kt * 32 + halfid * 8 + j;
        int k1 = kt * 32 + 16 + halfid * 8 + j;
        dst[j]     = (_Float16)W[(size_t)k0 * N + n];
        dst[j + 8] = (_Float16)W[(size_t)k1 * N + n];
    }
}

// ---------------------------------------------------------------------------
// C[M x N] = A[M x 256] * W + bias.  A is f16 row-major in global memory.
// Block = 256 threads (8 waves), M tile = 64 rows, async-DMA'd to LDS
// (global_load_async_to_lds_b128 / ASYNCcnt).  Each wave owns NT column
// tiles processed in ONE pass over kt: each A fragment is loaded from LDS
// once and immediately consumed by NT WMMAs; each B fragment is reused by
// 4 M sub-tiles.  No loop-invariant fragment sets -> no spills.
// ---------------------------------------------------------------------------
template <int RELU, int OUT16, int NT>
__global__ __launch_bounds__(256)
void wmma_gemm64_kernel(const _Float16* __restrict__ Ah,
                        const _Float16* __restrict__ Wp,
                        const float* __restrict__ bias,
                        float* __restrict__ Cf,
                        _Float16* __restrict__ Ch,
                        int M, int N) {
    __shared__ __align__(16) _Float16 lds_a[64][264];  // 264 = 256 + 8 pad halves
    const int tid = threadIdx.x;
    const int m0  = blockIdx.x * 64;

    // --- async DMA stage: 64 rows x 512B, 16B per lane, 8 rows per wave ----
    {
        int col16 = tid & 31;
        int rbase = tid >> 5;
        uint32_t lds_base = (uint32_t)(uintptr_t)(&lds_a[0][0]);
#pragma unroll
        for (int i = 0; i < 8; ++i) {
            int row = i * 8 + rbase;
            int gr  = m0 + row;
            if (gr >= M) gr = M - 1;  // clamp; tail masked at store
            const void* gp = (const void*)(Ah + (size_t)gr * KDIM + col16 * 8);
            uint32_t lp = lds_base + (uint32_t)(row * 528 + col16 * 16);
            asm volatile("global_load_async_to_lds_b128 %0, %1, off"
                         :: "v"(lp), "v"(gp) : "memory");
        }
    }
    asm volatile("s_wait_asynccnt 0x0" ::: "memory");
    __syncthreads();

    const int lane   = tid & 31;
    const int wv     = tid >> 5;     // wave id = first column tile
    const int r      = lane & 15;    // A row within sub-tile / C column
    const int halfid = lane >> 4;

    v8f acc[NT][4];
#pragma unroll
    for (int t = 0; t < NT; ++t) {
        float bv = bias[(wv + 8 * t) * 16 + r];
#pragma unroll
        for (int mi = 0; mi < 4; ++mi)
#pragma unroll
            for (int i = 0; i < 8; ++i) acc[t][mi][i] = bv;
    }

#pragma unroll
    for (int kt = 0; kt < 8; ++kt) {
        v16h b[NT];
#pragma unroll
        for (int t = 0; t < NT; ++t) {
            const _Float16* wp =
                Wp + ((((size_t)(wv + 8 * t) * 8 + kt) * 32 + lane) * 16);
            v8h blo = *(const v8h*)(wp);
            v8h bhi = *(const v8h*)(wp + 8);
#pragma unroll
            for (int i = 0; i < 8; ++i) { b[t][i] = blo[i]; b[t][i + 8] = bhi[i]; }
        }
#pragma unroll
        for (int mi = 0; mi < 4; ++mi) {
            const _Float16* ap = &lds_a[mi * 16 + r][kt * 32 + halfid * 8];
            v8h alo = *(const v8h*)ap;
            v8h ahi = *(const v8h*)(ap + 16);
            v16h a;
#pragma unroll
            for (int i = 0; i < 8; ++i) { a[i] = alo[i]; a[i + 8] = ahi[i]; }
#pragma unroll
            for (int t = 0; t < NT; ++t) {
                acc[t][mi] = __builtin_amdgcn_wmma_f32_16x16x32_f16(
                    false, a, false, b[t], (short)0, acc[t][mi], false, false);
            }
        }
    }

    // --- store: fast path for full tiles (uniform branch) ------------------
    if (m0 + 64 <= M) {
#pragma unroll
        for (int t = 0; t < NT; ++t) {
            int n0 = (wv + 8 * t) * 16 + r;
#pragma unroll
            for (int mi = 0; mi < 4; ++mi)
#pragma unroll
                for (int v = 0; v < 8; ++v) {
                    float o = acc[t][mi][v];
                    if (RELU) o = fmaxf(o, 0.0f);
                    size_t gm = (size_t)(m0 + mi * 16 + v + 8 * halfid);
                    if (OUT16) Ch[gm * N + n0] = (_Float16)o;
                    else       Cf[gm * N + n0] = o;
                }
        }
    } else {
#pragma unroll
        for (int t = 0; t < NT; ++t) {
            int n0 = (wv + 8 * t) * 16 + r;
#pragma unroll
            for (int mi = 0; mi < 4; ++mi)
#pragma unroll
                for (int v = 0; v < 8; ++v) {
                    float o = acc[t][mi][v];
                    if (RELU) o = fmaxf(o, 0.0f);
                    int gm = m0 + mi * 16 + v + 8 * halfid;
                    if (gm < M) {
                        if (OUT16) Ch[(size_t)gm * N + n0] = (_Float16)o;
                        else       Cf[(size_t)gm * N + n0] = o;
                    }
                }
        }
    }
}

// ---------------------------------------------------------------------------
// Deformable sampling: one wave per (query, head); lane = channel (HD=32).
// value is f16 (64B coalesced gathers); accumulate f32; emit f16 (consumed
// only by the out-projection GEMM).
// ---------------------------------------------------------------------------
__global__ void dfa_sample_kernel(const _Float16* __restrict__ value,
                                  const float* __restrict__ off,
                                  const float* __restrict__ attn,
                                  const float* __restrict__ ref,
                                  const int* __restrict__ shapes,
                                  const int* __restrict__ starts,
                                  _Float16* __restrict__ out, int Lq) {
    const int q    = blockIdx.x;
    const int head = threadIdx.x >> 5;
    const int lane = threadIdx.x & 31;

    float logit = (lane < 16) ? attn[(size_t)q * 128 + head * 16 + lane] : -1e30f;
    float m = logit;
#pragma unroll
    for (int o = 8; o >= 1; o >>= 1) m = fmaxf(m, __shfl_xor(m, o, 32));
    float e = (lane < 16) ? __expf(logit - m) : 0.0f;
    float s = e;
#pragma unroll
    for (int o = 8; o >= 1; o >>= 1) s += __shfl_xor(s, o, 32);
    float aw = e / s;

    float offv = off[(size_t)q * 256 + head * 32 + lane];  // lane = (l*4+p)*2+xy

    float acc = 0.0f;
#pragma unroll
    for (int l = 0; l < 4; ++l) {
        const int Hl = shapes[2 * l + 0];
        const int Wl = shapes[2 * l + 1];
        const int st = starts[l];
        const float fWl = (float)Wl, fHl = (float)Hl;
        const float refx = ref[(size_t)q * 8 + l * 2 + 0];
        const float refy = ref[(size_t)q * 8 + l * 2 + 1];
#pragma unroll
        for (int p = 0; p < 4; ++p) {
            int i = l * 4 + p;
            float ox  = __shfl(offv, 2 * i, 32);
            float oy  = __shfl(offv, 2 * i + 1, 32);
            float w_a = __shfl(aw, i, 32);
            float x = refx * fWl + ox - 0.5f;   // (ref + off/W)*W - 0.5
            float y = refy * fHl + oy - 0.5f;
            float x0 = floorf(x), y0 = floorf(y);
            float fx = x - x0, fy = y - y0;
#pragma unroll
            for (int dy = 0; dy < 2; ++dy) {
#pragma unroll
                for (int dx = 0; dx < 2; ++dx) {
                    float xi = x0 + (float)dx, yi = y0 + (float)dy;
                    float w  = (dx ? fx : 1.0f - fx) * (dy ? fy : 1.0f - fy);
                    bool valid = (xi >= 0.0f) && (xi < fWl) && (yi >= 0.0f) && (yi < fHl);
                    int xc = (int)fminf(fmaxf(xi, 0.0f), fWl - 1.0f);
                    int yc = (int)fminf(fmaxf(yi, 0.0f), fHl - 1.0f);
                    size_t idx = (size_t)(st + yc * Wl + xc) * 256 + head * 32 + lane;
                    float g = (float)value[idx];
                    acc += valid ? (w * w_a * g) : 0.0f;
                }
            }
        }
    }
    out[(size_t)q * 256 + head * 32 + lane] = (_Float16)acc;
}

// ---------------------------------------------------------------------------
// out = LayerNorm(a + b) * g + beta; one wave per 256-channel row.
// Optionally emits an f16 mirror for the next GEMM's A operand.
// ---------------------------------------------------------------------------
__global__ void add_ln_kernel(const float* __restrict__ a, const float* __restrict__ b,
                              const float* __restrict__ g, const float* __restrict__ be,
                              float* __restrict__ out, _Float16* __restrict__ outh, int M) {
    int row  = blockIdx.x * 8 + (threadIdx.x >> 5);
    int lane = threadIdx.x & 31;
    if (row >= M) return;
    float v[8];
    float sum = 0.0f;
#pragma unroll
    for (int j = 0; j < 8; ++j) {
        int c = lane + 32 * j;
        v[j] = a[(size_t)row * 256 + c] + b[(size_t)row * 256 + c];
        sum += v[j];
    }
#pragma unroll
    for (int o = 16; o >= 1; o >>= 1) sum += __shfl_xor(sum, o, 32);
    float mean = sum * (1.0f / 256.0f);
    float var = 0.0f;
#pragma unroll
    for (int j = 0; j < 8; ++j) { float d = v[j] - mean; var += d * d; }
#pragma unroll
    for (int o = 16; o >= 1; o >>= 1) var += __shfl_xor(var, o, 32);
    var *= (1.0f / 256.0f);
    float rstd = __frsqrt_rn(var + 1e-5f);
#pragma unroll
    for (int j = 0; j < 8; ++j) {
        int c = lane + 32 * j;
        float o = (v[j] - mean) * rstd * g[c] + be[c];
        out[(size_t)row * 256 + c] = o;
        if (outh) outh[(size_t)row * 256 + c] = (_Float16)o;
    }
}

// q_h = f16(query + pos); query_h = f16(query)
__global__ void add_h_kernel(const float* __restrict__ a, const float* __restrict__ b,
                             _Float16* __restrict__ qh, _Float16* __restrict__ ah, size_t n) {
    size_t i = (size_t)blockIdx.x * blockDim.x + threadIdx.x;
    if (i < n) {
        float av = a[i];
        qh[i] = (_Float16)(av + b[i]);
        ah[i] = (_Float16)av;
    }
}

// ---------------------------------------------------------------------------
extern "C" void kernel_launch(void* const* d_in, const int* in_sizes, int n_in,
                              void* d_out, int out_size, void* d_ws, size_t ws_size,
                              hipStream_t stream) {
    const float* query  = (const float*)d_in[0];
    const float* ref    = (const float*)d_in[1];
    const float* pos    = (const float*)d_in[2];
    const int*   shapes = (const int*)d_in[3];
    const int*   starts = (const int*)d_in[4];
    const float* w_off  = (const float*)d_in[5];
    const float* b_off  = (const float*)d_in[6];
    const float* w_attn = (const float*)d_in[7];
    const float* b_attn = (const float*)d_in[8];
    const float* w_val  = (const float*)d_in[9];
    const float* b_val  = (const float*)d_in[10];
    const float* w_out  = (const float*)d_in[11];
    const float* b_out  = (const float*)d_in[12];
    const float* g1     = (const float*)d_in[13];
    const float* beta1  = (const float*)d_in[14];
    const float* w1     = (const float*)d_in[15];
    const float* b1     = (const float*)d_in[16];
    const float* w2     = (const float*)d_in[17];
    const float* b2     = (const float*)d_in[18];
    const float* g2     = (const float*)d_in[19];
    const float* beta2  = (const float*)d_in[20];

    const int Lq = in_sizes[0] / 256;  // 43054
    const size_t F = (size_t)Lq * 256;

    // byte-aliased workspace regions (phases never overlap in time)
    char* p = (char*)d_ws;
    float*    R1 = (float*)p;    p += F * 4;  // off          -> x
    float*    R2 = (float*)p;    p += F * 4;  // attn logits  -> attn_out -> ffn
    _Float16* R3 = (_Float16*)p; p += F * 2;  // query_h      -> acc_h
    _Float16* R4 = (_Float16*)p; p += F * 2;  // q_h          -> x_h
    _Float16* R5 = (_Float16*)p; p += F * 2;  // value_h      -> ffn_hidden_h
    _Float16* wv_p = (_Float16*)p;
    _Float16* wo_p = wv_p + 65536;
    _Float16* wa_p = wo_p + 65536;
    _Float16* wu_p = wa_p + 32768;
    _Float16* w1_p = wu_p + 65536;
    _Float16* w2_p = w1_p + 65536;

    // pack weights into WMMA B-fragment layout (f16, L2-resident)
    pack_w_kernel<<<16, 256, 0, stream>>>(w_val,  wv_p, 256);
    pack_w_kernel<<<16, 256, 0, stream>>>(w_off,  wo_p, 256);
    pack_w_kernel<<<8,  256, 0, stream>>>(w_attn, wa_p, 128);
    pack_w_kernel<<<16, 256, 0, stream>>>(w_out,  wu_p, 256);
    pack_w_kernel<<<16, 256, 0, stream>>>(w1,     w1_p, 256);
    pack_w_kernel<<<16, 256, 0, stream>>>(w2,     w2_p, 256);

    add_h_kernel<<<(unsigned)((F + 255) / 256), 256, 0, stream>>>(query, pos, R4, R3, F);

    const int mb = (Lq + 63) / 64;
    // projections (A operands are f16)
    wmma_gemm64_kernel<0, 1, 2><<<mb, 256, 0, stream>>>(R3, wv_p, b_val,  nullptr, R5, Lq, 256); // value (f16)
    wmma_gemm64_kernel<0, 0, 2><<<mb, 256, 0, stream>>>(R4, wo_p, b_off,  R1, nullptr, Lq, 256); // offsets
    wmma_gemm64_kernel<0, 0, 1><<<mb, 256, 0, stream>>>(R4, wa_p, b_attn, R2, nullptr, Lq, 128); // attn logits

    // deformable sampling -> acc (f16, head-major channels)
    dfa_sample_kernel<<<Lq, 256, 0, stream>>>(R5, R1, R2, ref, shapes, starts, R3, Lq);

    // output projection + residual LN (x kept f32 for final residual, + f16 mirror)
    wmma_gemm64_kernel<0, 0, 2><<<mb, 256, 0, stream>>>(R3, wu_p, b_out, R2, nullptr, Lq, 256);
    add_ln_kernel<<<(Lq + 7) / 8, 256, 0, stream>>>(query, R2, g1, beta1, R1, R4, Lq);

    // FFN
    wmma_gemm64_kernel<1, 1, 2><<<mb, 256, 0, stream>>>(R4, w1_p, b1, nullptr, R5, Lq, 256); // relu, f16
    wmma_gemm64_kernel<0, 0, 2><<<mb, 256, 0, stream>>>(R5, w2_p, b2, R2, nullptr, Lq, 256);

    add_ln_kernel<<<(Lq + 7) / 8, 256, 0, stream>>>(R1, R2, g2, beta2, (float*)d_out, nullptr, Lq);
}
